// AttentionModel_22677427323379
// MI455X (gfx1250) — matrix operand order repaired and checked by reading, not compiled, not run on hardware
//
#include <hip/hip_runtime.h>

// ---------------- model dims ----------------
#define B_    256
#define IH_   200
#define LH_   100
#define G_    301            // IH + LH + 1
#define E_    256
#define FFH_  512
#define MT_   (B_*G_)        // 77056 rows, multiple of 16 and of 128
#define GPAD  328            // padded token count for score/prob LDS rows (bank-friendly)
#define VTP   320            // padded token count for transposed V (10 K-blocks of 32)

typedef __attribute__((ext_vector_type(8)))  float  v8f;
typedef __attribute__((ext_vector_type(16))) __bf16 v16bf;
typedef __attribute__((ext_vector_type(8)))  __bf16 v8bf;

union BFrag { v16bf v; v8bf h[2]; };

static __device__ __forceinline__ v8bf ld8(const __bf16* p){ return *(const v8bf*)p; }

// =====================================================================
// Generic bf16 WMMA GEMM: C[M,ND] = act(A[M,KD] @ W[KD,ND] + bias) (+res)
// A bf16 row-major, W fp32 row-major (transposed into LDS as bf16).
// 256 threads = 8 waves, each wave owns 16 rows; block tile = 128 rows.
// A fragments fully preloaded; B fragments double-buffered across n-tiles
// so each tile is one LDS clause + an uninterrupted WMMA chain.
// OMODE bits: 1 = write fp32 row-major, 2 = write bf16 row-major,
//             4 = write bf16 transposed per-batch [b][col][g] with VTP pad.
// =====================================================================
template<int KD, int ND, bool BIAS, int ACT, bool RES, int OMODE>
__global__ __launch_bounds__(256) void gemm_ws(
    const __bf16* __restrict__ A, const float* __restrict__ W,
    const float* __restrict__ bias, const float* __restrict__ resid,
    float* __restrict__ Cf, __bf16* __restrict__ Cb)
{
  constexpr int KS  = KD + 8;                // padded K stride in LDS
  constexpr int NKB = KD / 32;
  constexpr int NT  = ND / 16;               // even for all configs used
  extern __shared__ __bf16 wl[];             // ND * KS bf16, wl[n*KS + k]
  const int tid = threadIdx.x;
  for (int idx = tid; idx < KD * ND; idx += 256) {
    int k = idx / ND, n = idx - k * ND;
    wl[n * KS + k] = (__bf16)W[idx];
  }
  __syncthreads();

  const int wave = tid >> 5, lane = tid & 31;
  const int row  = lane & 15, half = lane >> 4;
  const int m0   = blockIdx.x * 128 + wave * 16;
  const __bf16* arow = A + (long)(m0 + row) * KD + half * 8;

  BFrag af[NKB];
#pragma unroll
  for (int kb = 0; kb < NKB; ++kb) {
    af[kb].h[0] = ld8(arow + kb * 32);
    af[kb].h[1] = ld8(arow + kb * 32 + 16);
  }

  auto loadB = [&](BFrag (&dst)[NKB], int nt) {
    const __bf16* bbase = wl + (nt * 16 + row) * KS + half * 8;
#pragma unroll
    for (int kb = 0; kb < NKB; ++kb) {
      dst[kb].h[0] = *(const v8bf*)(bbase + kb * 32);
      dst[kb].h[1] = *(const v8bf*)(bbase + kb * 32 + 16);
    }
  };
  auto compute = [&](BFrag (&bs)[NKB], int nt) {
    v8f acc = {};
#pragma unroll
    for (int kb = 0; kb < NKB; ++kb)
      acc = __builtin_amdgcn_wmma_f32_16x16x32_bf16(
          false, af[kb].v, false, bs[kb].v, (short)0, acc, false, false);
    const int col = nt * 16 + row;
    const float bv = BIAS ? bias[col] : 0.f;
#pragma unroll
    for (int v = 0; v < 8; ++v) {
      const int r = m0 + v + half * 8;
      float x = acc[v] + bv;
      if (ACT == 1) x = fmaxf(x, 0.f);
      if (ACT == 2) x = (x > 0.f) ? x : 0.01f * x;
      if (RES)      x += resid[(long)r * ND + col];
      if (OMODE & 1) Cf[(long)r * ND + col] = x;
      if (OMODE & 2) Cb[(long)r * ND + col] = (__bf16)x;
      if (OMODE & 4) { const int bb = r / G_, g = r - bb * G_;
                       Cb[((long)bb * E_ + col) * VTP + g] = (__bf16)x; }
    }
  };

  BFrag b0[NKB], b1[NKB];
  loadB(b0, 0);
#pragma unroll 1
  for (int nt = 0; nt < NT; nt += 2) {
    loadB(b1, nt + 1);                       // prefetch while computing b0
    compute(b0, nt);
    if (nt + 2 < NT) loadB(b0, nt + 2);      // prefetch while computing b1
    compute(b1, nt + 1);
  }
}

// =====================================================================
// Fused attention: one wave per (batch, 16-query tile).
// scores = Q K^T * 1/16 + mask bias ; softmax ; out = P V  (all WMMA bf16)
// Mask bias precomputed into LDS (no global loads / drains in score loop);
// K tiles and V tiles double-buffered; P fragments hoisted out of loop.
// =====================================================================
__global__ __launch_bounds__(32) void attn_fused(
    const __bf16* __restrict__ Qb, const __bf16* __restrict__ Kb,
    const __bf16* __restrict__ Vt, const float* __restrict__ obs,
    __bf16* __restrict__ AVb)
{
  __shared__ float  sc[16 * GPAD];
  __shared__ float  biasS[VTP];
  __shared__ __bf16 pb[16 * GPAD];
  const int b = blockIdx.y, qt = blockIdx.x;
  const int lane = threadIdx.x, row = lane & 15, half = lane >> 4;
  const long base = (long)b * G_;
  const int q0 = qt * 16;
  const int qrow = min(q0 + row, G_ - 1);

  // precompute attention bias per token: -(1-mask)*1e9 for real tokens,
  // -1e30 for padding -> score loop needs no global loads or branches.
  for (int t0 = lane; t0 < VTP; t0 += 32) {
    float bv = -1e30f;
    if (t0 < G_) bv = (1.f - obs[(base + t0) * 9 + 8]) * -1e9f;
    biasS[t0] = bv;
  }

  const __bf16* qp = Qb + (base + qrow) * E_ + half * 8;
  BFrag qf[8];
#pragma unroll
  for (int kb = 0; kb < 8; ++kb) {
    qf[kb].h[0] = ld8(qp + kb * 32);
    qf[kb].h[1] = ld8(qp + kb * 32 + 16);
  }
  __syncthreads();

  // ---- scores over 20 key tiles (320 padded tokens), double-buffered ----
  auto loadK = [&](BFrag (&dst)[8], int kt) {
    const int krow = min(kt * 16 + row, G_ - 1);
    const __bf16* kp = Kb + (base + krow) * E_ + half * 8;
#pragma unroll
    for (int kb = 0; kb < 8; ++kb) {
      dst[kb].h[0] = ld8(kp + kb * 32);
      dst[kb].h[1] = ld8(kp + kb * 32 + 16);
    }
  };
  auto scoreT = [&](BFrag (&ks)[8], int kt) {
    v8f acc = {};
#pragma unroll
    for (int kb = 0; kb < 8; ++kb)
      acc = __builtin_amdgcn_wmma_f32_16x16x32_bf16(
          false, qf[kb].v, false, ks[kb].v, (short)0, acc, false, false);
    const int tok = kt * 16 + row;
    const float biasv = biasS[tok];          // -1e30 on padding dominates
#pragma unroll
    for (int v = 0; v < 8; ++v)
      sc[(v + half * 8) * GPAD + tok] = acc[v] * 0.0625f + biasv;
  };
  {
    BFrag k0[8], k1[8];
    loadK(k0, 0);
#pragma unroll 1
    for (int kt = 0; kt < 20; kt += 2) {
      loadK(k1, kt + 1);
      scoreT(k0, kt);
      if (kt + 2 < 20) loadK(k0, kt + 2);
      scoreT(k1, kt + 1);
    }
  }
  __syncthreads();

  // ---- per-row softmax, re-encode probs as bf16 A-operand ----
  if (lane < 16) {
    float* srow = sc + lane * GPAD;
    float m = -1e30f;
    for (int j = 0; j < G_; ++j) m = fmaxf(m, srow[j]);
    float s = 0.f;
    for (int j = 0; j < G_; ++j) { const float e = __expf(srow[j] - m); srow[j] = e; s += e; }
    const float inv = 1.f / (s + 1e-20f);
    __bf16* prow = pb + lane * GPAD;
    for (int j = 0; j < G_; ++j)  prow[j] = (__bf16)(srow[j] * inv);
    for (int j = G_; j < GPAD; ++j) prow[j] = (__bf16)0.f;
  }
  __syncthreads();

  // ---- out = P (16x320) @ V (320xE): P fragments hoisted, V double-buffered ----
  const __bf16* prow = pb + row * GPAD + half * 8;
  BFrag pf[10];
#pragma unroll
  for (int kb = 0; kb < 10; ++kb) {
    pf[kb].h[0] = *(const v8bf*)(prow + kb * 32);
    pf[kb].h[1] = *(const v8bf*)(prow + kb * 32 + 16);
  }
  const long vb0 = (long)b * E_ * VTP;
  auto loadV = [&](BFrag (&dst)[10], int nt) {
    const __bf16* vp = Vt + vb0 + (long)(nt * 16 + row) * VTP + half * 8;
#pragma unroll
    for (int kb = 0; kb < 10; ++kb) {
      dst[kb].h[0] = ld8(vp + kb * 32);
      dst[kb].h[1] = ld8(vp + kb * 32 + 16);
    }
  };
  auto avT = [&](BFrag (&vs)[10], int nt) {
    v8f acc = {};
#pragma unroll
    for (int kb = 0; kb < 10; ++kb)
      acc = __builtin_amdgcn_wmma_f32_16x16x32_bf16(
          false, pf[kb].v, false, vs[kb].v, (short)0, acc, false, false);
#pragma unroll
    for (int v = 0; v < 8; ++v) {
      const int q = q0 + v + half * 8;
      if (q < G_) AVb[(base + q) * E_ + nt * 16 + row] = (__bf16)acc[v];
    }
  };
  {
    BFrag v0[10], v1[10];
    loadV(v0, 0);
#pragma unroll 1
    for (int nt = 0; nt < 16; nt += 2) {
      loadV(v1, nt + 1);
      avT(v0, nt);
      if (nt + 2 < 16) loadV(v0, nt + 2);
      avT(v1, nt + 1);
    }
  }
}

// =====================================================================
// Token embedding MLPs (internal/leaf/next), W2 staged in LDS.
// =====================================================================
__global__ __launch_bounds__(256) void embed_kernel(
    const float* __restrict__ obs,
    const float* __restrict__ wi1, const float* __restrict__ bi1,
    const float* __restrict__ wi2, const float* __restrict__ bi2,
    const float* __restrict__ wl1, const float* __restrict__ bl1,
    const float* __restrict__ wl2, const float* __restrict__ bl2,
    const float* __restrict__ wn1, const float* __restrict__ bn1,
    const float* __restrict__ wn2, const float* __restrict__ bn2,
    float* __restrict__ H, __bf16* __restrict__ Hb)
{
  extern __shared__ float sm[];   // 3*8192 weights + 3*256 biases
  const int tid = threadIdx.x;
  for (int i = tid; i < 8192; i += 256) {
    sm[i] = wi2[i]; sm[8192 + i] = wl2[i]; sm[16384 + i] = wn2[i];
  }
  if (tid < 256) { sm[24576 + tid] = bi2[tid]; sm[24832 + tid] = bl2[tid]; sm[25088 + tid] = bn2[tid]; }
  __syncthreads();

  const int t = blockIdx.x * 256 + tid;     // exactly MT_ threads total
  const int g = t % G_;
  int which, nin; const float *w1, *b1;
  if (g < IH_)              { which = 0; nin = 6; w1 = wi1; b1 = bi1; }
  else if (g < IH_ + LH_)   { which = 1; nin = 8; w1 = wl1; b1 = bl1; }
  else                      { which = 2; nin = 6; w1 = wn1; b1 = bn1; }

  float x[8];
  for (int i = 0; i < 8; ++i) x[i] = (i < nin) ? obs[(long)t * 9 + i] : 0.f;

  float hid[32];
#pragma unroll
  for (int j = 0; j < 32; ++j) {
    float a = b1[j];
    for (int i = 0; i < 8; ++i) if (i < nin) a += x[i] * w1[i * 32 + j];
    hid[j] = (a > 0.f) ? a : 0.01f * a;     // leaky_relu 0.01
  }
  const float* w2 = sm + which * 8192;
  const float* b2 = sm + 24576 + which * 256;
  float*  hr  = H  + (long)t * E_;
  __bf16* hbr = Hb + (long)t * E_;
  for (int e = 0; e < E_; ++e) {
    float a = b2[e];
#pragma unroll
    for (int j = 0; j < 32; ++j) a += hid[j] * w2[j * 256 + e];
    hr[e] = a; hbr[e] = (__bf16)a;
  }
}

// =====================================================================
// LayerNorm (no affine): one wave per 256-wide row; writes fp32 + bf16.
// =====================================================================
__global__ __launch_bounds__(256) void ln_kernel(
    const float* __restrict__ X, float* __restrict__ H, __bf16* __restrict__ Hb)
{
  const int wave = threadIdx.x >> 5, lane = threadIdx.x & 31;
  const long r = (long)blockIdx.x * 8 + wave;
  const float* x = X + r * E_;
  float v[8], s = 0.f, s2 = 0.f;
#pragma unroll
  for (int i = 0; i < 8; ++i) { const float t = x[lane + i * 32]; v[i] = t; s += t; s2 += t * t; }
#pragma unroll
  for (int m = 16; m >= 1; m >>= 1) { s += __shfl_xor(s, m); s2 += __shfl_xor(s2, m); }
  const float mean = s * (1.f / E_);
  const float var  = s2 * (1.f / E_) - mean * mean;
  const float rs   = rsqrtf(var + 1e-5f);
  float*  hr  = H  + r * E_;
  __bf16* hbr = Hb + r * E_;
#pragma unroll
  for (int i = 0; i < 8; ++i) {
    const float y = (v[i] - mean) * rs;
    hr[lane + i * 32] = y; hbr[lane + i * 32] = (__bf16)y;
  }
}

// =====================================================================
// Pointer head: masked mean -> fc -> u = Wpn[:, :E] @ fc -> leaf logits.
// One block per batch element.
// =====================================================================
__global__ __launch_bounds__(256) void pointer_kernel(
    const float* __restrict__ H, const float* __restrict__ obs,
    const float* __restrict__ w_pn, const float* __restrict__ w_fc,
    float* __restrict__ out)
{
  __shared__ float maskS[G_];
  __shared__ float geS[E_], fcS[E_], uS[E_];
  __shared__ float lg[LH_];
  __shared__ float vlS;
  const int b = blockIdx.x, t = threadIdx.x;
  const long base = (long)b * G_;
  for (int g = t; g < G_; g += 256) maskS[g] = obs[(base + g) * 9 + 8];
  __syncthreads();
  if (t == 0) { float s = 0.f; for (int g = 0; g < G_; ++g) s += maskS[g]; vlS = s; }
  __syncthreads();
  float acc = 0.f;
  for (int g = 0; g < G_; ++g) acc += H[(base + g) * E_ + t] * maskS[g];
  geS[t] = acc / vlS;
  __syncthreads();
  float f = 0.f;
  for (int d = 0; d < E_; ++d) f += geS[d] * w_fc[d * E_ + t];
  fcS[t] = f;
  __syncthreads();
  float u = 0.f;
  for (int e = 0; e < E_; ++e) u += w_pn[t * (3 * E_) + e] * fcS[e];
  uS[t] = u;
  __syncthreads();
  if (t < LH_) {
    const float* hr = H + (base + IH_ + t) * E_;
    float c = 0.f;
    for (int e = 0; e < E_; ++e) c += hr[e] * uS[e];
    c *= maskS[IH_ + t];                       // trans = h * mask
    lg[t] = tanhf(c * 0.0625f) * 10.f;         // tanh(compat*scale)*CLIP
  }
  __syncthreads();
  if (t == 0) {
    float m = -1e30f;
    for (int i = 0; i < LH_; ++i) m = fmaxf(m, lg[i]);
    float s = 0.f;
    for (int i = 0; i < LH_; ++i) { const float e = __expf(lg[i] - m); lg[i] = e; s += e; }
    float s2 = 0.f;
    for (int i = 0; i < LH_; ++i) {
      const float p = lg[i] / s;
      const float mk = p * maskS[IH_ + i] + 1e-20f;
      lg[i] = mk; s2 += mk;
    }
    for (int i = 0; i < LH_; ++i) out[(long)b * LH_ + i] = lg[i] / s2;
  }
}

// =====================================================================
extern "C" void kernel_launch(void* const* d_in, const int* in_sizes, int n_in,
                              void* d_out, int out_size, void* d_ws, size_t ws_size,
                              hipStream_t stream)
{
  (void)in_sizes; (void)n_in; (void)out_size; (void)ws_size;
  const float* obs    = (const float*)d_in[0];
  const float* wi1    = (const float*)d_in[1];
  const float* bi1    = (const float*)d_in[2];
  const float* wi2    = (const float*)d_in[3];
  const float* bi2    = (const float*)d_in[4];
  const float* wl1    = (const float*)d_in[5];
  const float* bl1    = (const float*)d_in[6];
  const float* wl2    = (const float*)d_in[7];
  const float* bl2    = (const float*)d_in[8];
  const float* wn1    = (const float*)d_in[9];
  const float* bn1    = (const float*)d_in[10];
  const float* wn2    = (const float*)d_in[11];
  const float* bn2    = (const float*)d_in[12];
  const float* enc_wq = (const float*)d_in[13];
  const float* enc_wk = (const float*)d_in[14];
  const float* enc_wv = (const float*)d_in[15];
  const float* enc_wo = (const float*)d_in[16];
  const float* enc_wf1= (const float*)d_in[17];
  const float* enc_bf1= (const float*)d_in[18];
  const float* enc_wf2= (const float*)d_in[19];
  const float* enc_bf2= (const float*)d_in[20];
  const float* w_pn   = (const float*)d_in[21];
  const float* w_fc   = (const float*)d_in[22];

  // workspace layout
  float*  h    = (float*)d_ws;                          // MT_*E_ fp32
  float*  tmp  = h + (size_t)MT_ * E_;                  // MT_*E_ fp32
  __bf16* hb   = (__bf16*)(tmp + (size_t)MT_ * E_);     // MT_*E_ bf16
  __bf16* qb   = hb + (size_t)MT_ * E_;                 // MT_*E_ bf16
  __bf16* kbuf = qb + (size_t)MT_ * E_;                 // MT_*E_ bf16
  __bf16* vt   = kbuf + (size_t)MT_ * E_;               // B_*E_*VTP bf16 (transposed V)
  __bf16* avb  = vt + (size_t)B_ * E_ * VTP;            // MT_*E_ bf16
  __bf16* ffb  = qb;                                    // MT_*FFH bf16, reuses qb+kbuf

  const int    GB    = MT_ / 128;                       // 602 GEMM blocks
  const size_t sh256 = (size_t)E_  * (E_  + 8) * 2;     // 135168
  const size_t sh512 = (size_t)FFH_* (E_  + 8) * 2;     // 270336
  const size_t shK512= (size_t)E_  * (FFH_+ 8) * 2;     // 266240
  const size_t shEmb = (3 * 8192 + 3 * 256) * sizeof(float);

  embed_kernel<<<G_, 256, shEmb, stream>>>(obs, wi1, bi1, wi2, bi2, wl1, bl1, wl2, bl2,
                                           wn1, bn1, wn2, bn2, h, hb);

  for (int l = 0; l < 2; ++l) {
    const float* Wq  = enc_wq  + (size_t)l * E_ * E_;
    const float* Wk  = enc_wk  + (size_t)l * E_ * E_;
    const float* Wv  = enc_wv  + (size_t)l * E_ * E_;
    const float* Wo  = enc_wo  + (size_t)l * E_ * E_;
    const float* Wf1 = enc_wf1 + (size_t)l * E_ * FFH_;
    const float* Bf1 = enc_bf1 + (size_t)l * FFH_;
    const float* Wf2 = enc_wf2 + (size_t)l * FFH_ * E_;
    const float* Bf2 = enc_bf2 + (size_t)l * E_;

    gemm_ws<256,256,false,0,false,2><<<GB,256,sh256,stream>>>(hb, Wq, nullptr, nullptr, nullptr, qb);
    gemm_ws<256,256,false,0,false,2><<<GB,256,sh256,stream>>>(hb, Wk, nullptr, nullptr, nullptr, kbuf);
    gemm_ws<256,256,false,0,false,4><<<GB,256,sh256,stream>>>(hb, Wv, nullptr, nullptr, nullptr, vt);

    attn_fused<<<dim3(19, B_), 32, 0, stream>>>(qb, kbuf, vt, obs, avb);

    gemm_ws<256,256,false,0,true,1><<<GB,256,sh256,stream>>>(avb, Wo, nullptr, h, tmp, nullptr);
    ln_kernel<<<MT_/8, 256, 0, stream>>>(tmp, h, hb);

    gemm_ws<256,512,true,1,false,2><<<GB,256,sh512,stream>>>(hb, Wf1, Bf1, nullptr, nullptr, ffb);
    gemm_ws<512,256,true,0,true,1><<<GB,256,shK512,stream>>>(ffb, Wf2, Bf2, h, tmp, nullptr);
    ln_kernel<<<MT_/8, 256, 0, stream>>>(tmp, h, hb);
  }

  pointer_kernel<<<B_, 256, 0, stream>>>(h, obs, w_pn, w_fc, (float*)d_out);
}
